// GPT_34978213659273
// MI455X (gfx1250) — matrix operand order repaired and checked by reading, not compiled, not run on hardware
//
#include <hip/hip_runtime.h>
#include <math.h>
#include <stdint.h>

namespace {
constexpr int B_ = 4, T_ = 1024, D_ = 1024, H_ = 16, HD_ = 64, L_ = 6, V_ = 50257;
constexpr int D3_ = 3 * D_, D4_ = 4 * D_;
constexpr int NTOK = B_ * T_;                 // 4096 rows in the residual stream
}

typedef __attribute__((ext_vector_type(16))) _Float16 v16h;
typedef __attribute__((ext_vector_type(8)))  _Float16 v8h;
typedef __attribute__((ext_vector_type(8)))  float    v8f;

// ---------------------------------------------------------------------------
// WMMA helpers (CDNA5 V_WMMA_F32_16X16X32_F16, wave32)
// ---------------------------------------------------------------------------
__device__ __forceinline__ v16h cat8(v8h lo, v8h hi) {
  return __builtin_shufflevector(lo, hi, 0,1,2,3,4,5,6,7,8,9,10,11,12,13,14,15);
}

// A fragment: 16x32 f16 tile, row-major source, row stride `ld` (halves).
// lane: m = lane&15 ; K blocks: [kh..kh+7] and [16+kh..16+kh+7], kh=(lane>>4)*8
__device__ __forceinline__ v16h load_a_frag(const _Float16* A0, int ld, int k0) {
  int lane = threadIdx.x & 31;
  const _Float16* p = A0 + (size_t)(lane & 15) * ld + k0 + ((lane >> 4) << 3);
  v8h lo = *(const v8h*)p;
  v8h hi = *(const v8h*)(p + 16);
  return cat8(lo, hi);
}

// B fragment: B[k,n] = W[n,k] with W row-major [N,K], row stride `ld`.
// lane: n = lane&15 ; K blocks: [kh..kh+7] and [16+kh..16+kh+7], kh=(lane>>4)*8
__device__ __forceinline__ v16h load_b_frag(const _Float16* W0, int ld, int k0) {
  int lane = threadIdx.x & 31;
  const _Float16* p = W0 + (size_t)(lane & 15) * ld + k0 + ((lane >> 4) << 3);
  v8h lo = *(const v8h*)p;
  v8h hi = *(const v8h*)(p + 16);
  return cat8(lo, hi);
}

// A fragment sourced from an f32 LDS tile (attention probabilities).
__device__ __forceinline__ v16h load_a_frag_lds_f32(const float* S, int ld, int k0) {
  int lane = threadIdx.x & 31;
  const float* p = S + (lane & 15) * ld + k0 + ((lane >> 4) << 3);
  v16h f;
#pragma unroll
  for (int j = 0; j < 8; ++j) f[j] = (_Float16)p[j];
#pragma unroll
  for (int j = 0; j < 8; ++j) f[8 + j] = (_Float16)p[16 + j];
  return f;
}

__device__ __forceinline__ v8f wmma32(v16h a, v16h b, v8f c) {
  return __builtin_amdgcn_wmma_f32_16x16x32_f16(false, a, false, b, (short)0, c, false, false);
}

// LDS byte offset of a generic pointer to __shared__ (aperture: addr[31:0] = LDS offset)
__device__ __forceinline__ uint32_t lds_off(const void* p) { return (uint32_t)(uintptr_t)p; }

// ---------------------------------------------------------------------------
// f32 -> f16 weight conversion (grid-stride)
// ---------------------------------------------------------------------------
__global__ void cvt_f32_f16_kernel(const float* __restrict__ in, _Float16* __restrict__ out, long n) {
  for (long i = (long)blockIdx.x * blockDim.x + threadIdx.x; i < n; i += (long)gridDim.x * blockDim.x)
    out[i] = (_Float16)in[i];
}

// ---------------------------------------------------------------------------
// V transpose: vt[b,h,d,t] = qkv16[b,t, 2D + h*HD + d]   (makes P@V B-frags contiguous)
// ---------------------------------------------------------------------------
__global__ void vtrans_kernel(const _Float16* __restrict__ qkv, _Float16* __restrict__ vt) {
  long n = (long)B_ * H_ * HD_ * T_;
  for (long i = (long)blockIdx.x * blockDim.x + threadIdx.x; i < n; i += (long)gridDim.x * blockDim.x) {
    int t = (int)(i & (T_ - 1));
    long r = i >> 10;                 // b*H*HD + h*HD + d
    int d = (int)(r & (HD_ - 1));
    long bh = r >> 6;                 // b*H + h
    int h = (int)(bh & (H_ - 1));
    int b = (int)(bh >> 4);
    vt[i] = qkv[(size_t)(b * T_ + t) * D3_ + 2 * D_ + h * HD_ + d];
  }
}

// ---------------------------------------------------------------------------
// Embedding: x[b,t,:] = wte[idx[b,t],:] + wpe[t,:]
// ---------------------------------------------------------------------------
__global__ void embed_kernel(const int* __restrict__ idx, const float* __restrict__ wte,
                             const float* __restrict__ wpe, float* __restrict__ x) {
  long n = (long)NTOK * D_;
  for (long i = (long)blockIdx.x * blockDim.x + threadIdx.x; i < n; i += (long)gridDim.x * blockDim.x) {
    int d = (int)(i & (D_ - 1));
    long bt = i >> 10;
    int t = (int)(bt & (T_ - 1));
    x[i] = wte[(size_t)idx[bt] * D_ + d] + wpe[(size_t)t * D_ + d];
  }
}

// ---------------------------------------------------------------------------
// LayerNorm over D=1024, one block per row; f16 output (compact stride D).
// ---------------------------------------------------------------------------
__global__ __launch_bounds__(256) void layernorm_f16_kernel(
    const float* __restrict__ x, const float* __restrict__ w, const float* __restrict__ bb,
    _Float16* __restrict__ out, long row_stride) {
  __shared__ float red[256];
  int row = blockIdx.x, tid = threadIdx.x;
  const float* xr = x + (size_t)row * row_stride;
  float v[4], s = 0.f;
#pragma unroll
  for (int i = 0; i < 4; ++i) { v[i] = xr[tid + i * 256]; s += v[i]; }
  red[tid] = s; __syncthreads();
  for (int off = 128; off > 0; off >>= 1) { if (tid < off) red[tid] += red[tid + off]; __syncthreads(); }
  float mean = red[0] * (1.f / D_);
  __syncthreads();
  s = 0.f;
#pragma unroll
  for (int i = 0; i < 4; ++i) { float d = v[i] - mean; s += d * d; }
  red[tid] = s; __syncthreads();
  for (int off = 128; off > 0; off >>= 1) { if (tid < off) red[tid] += red[tid + off]; __syncthreads(); }
  float rstd = rsqrtf(red[0] * (1.f / D_) + 1e-5f);
#pragma unroll
  for (int i = 0; i < 4; ++i) {
    int d = tid + i * 256;
    out[(size_t)row * D_ + d] = (_Float16)((v[i] - mean) * rstd * w[d] + bb[d]);
  }
}

// ---------------------------------------------------------------------------
// WMMA GEMM: C[M,N] = A16[M,K] @ W16[N,K]^T + bias (+opt GELU, +opt residual)
// Block = 8 waves -> 128x128 tile; each wave 32x64 via 2x4 wmma tiles
// (8 wmma per 12 b128 loads). Requires M%128==0, N%128==0, K%32==0.
// ---------------------------------------------------------------------------
__global__ __launch_bounds__(256) void wmma_gemm_kernel(
    const _Float16* __restrict__ A, const _Float16* __restrict__ W,
    const float* __restrict__ bias, float* __restrict__ outf,
    const float* __restrict__ resid, _Float16* __restrict__ outh,
    int N, int K, int gelu) {
  int lane = threadIdx.x & 31;
  int wv = threadIdx.x >> 5;
  int m0 = blockIdx.y * 128 + (wv >> 1) * 32;
  int n0 = blockIdx.x * 128 + (wv & 1) * 64;
  const _Float16* A0 = A + (size_t)m0 * K;
  const _Float16* A1 = A0 + (size_t)16 * K;
  const _Float16* W0 = W + (size_t)n0 * K;
  const _Float16* W1 = W0 + (size_t)16 * K;
  const _Float16* W2 = W0 + (size_t)32 * K;
  const _Float16* W3 = W0 + (size_t)48 * K;

  v8f c[2][4] = {};
#pragma unroll 2
  for (int k0 = 0; k0 < K; k0 += 32) {
    __builtin_prefetch(W0 + (size_t)(lane & 15) * K + k0 + 256, 0, 1);  // global_prefetch_b8
    v16h a0 = load_a_frag(A0, K, k0);
    v16h a1 = load_a_frag(A1, K, k0);
    v16h b0 = load_b_frag(W0, K, k0);
    v16h b1 = load_b_frag(W1, K, k0);
    v16h b2 = load_b_frag(W2, K, k0);
    v16h b3 = load_b_frag(W3, K, k0);
    c[0][0] = wmma32(a0, b0, c[0][0]);
    c[0][1] = wmma32(a0, b1, c[0][1]);
    c[0][2] = wmma32(a0, b2, c[0][2]);
    c[0][3] = wmma32(a0, b3, c[0][3]);
    c[1][0] = wmma32(a1, b0, c[1][0]);
    c[1][1] = wmma32(a1, b1, c[1][1]);
    c[1][2] = wmma32(a1, b2, c[1][2]);
    c[1][3] = wmma32(a1, b3, c[1][3]);
  }

#pragma unroll
  for (int i = 0; i < 2; ++i)
#pragma unroll
    for (int j = 0; j < 4; ++j) {
      int mt = m0 + i * 16 + ((lane >> 4) << 3);
      int nn = n0 + j * 16 + (lane & 15);
      float bv = bias[nn];
      v8f cv = c[i][j];
#pragma unroll
      for (int r = 0; r < 8; ++r) {
        size_t idx = (size_t)(mt + r) * N + nn;
        float val = cv[r] + bv;
        if (gelu) val = 0.5f * val * (1.f + erff(val * 0.70710678118654752f));
        if (resid) val += resid[idx];
        if (outf) outf[idx] = val;
        if (outh) outh[idx] = (_Float16)val;
      }
    }
}

// ---------------------------------------------------------------------------
// Attention: one workgroup (4 waves) per (q-tile of 16, head, batch).
// Q tile staged into LDS via GLOBAL_LOAD_ASYNC_TO_LDS_B128 (ASYNCcnt path);
// scores S[16 x T] in LDS (f32); softmax in place; P@V via WMMA with
// A-fragments from LDS and contiguous B-fragments from pre-transposed V.
// ---------------------------------------------------------------------------
__global__ __launch_bounds__(128) void attn_kernel(const _Float16* __restrict__ qkv,
                                                   const _Float16* __restrict__ vt,
                                                   _Float16* __restrict__ y16) {
  extern __shared__ char smem_raw[];
  float* sS   = (float*)smem_raw;            // 16*1024 f32 scores -> exp values
  float* sO   = sS + 16 * 1024;              // 4 waves * 16*64 partial outputs
  float* sRed = sO + 4 * 16 * 64;            // 16*8 reduction scratch
  float* sSt  = sRed + 16 * 8;               // [0..15] rowmax, [16..31] 1/rowsum
  _Float16* sQ = (_Float16*)(sSt + 32);      // 16 x 64 Q tile (2 KB)

  int q0 = blockIdx.x * 16, h = blockIdx.y, b = blockIdx.z;
  int tid = threadIdx.x, lane = tid & 31, wave = tid >> 5;
  const float scale = 0.125f;  // 1/sqrt(64)

  // ---- Stage Q tile into LDS asynchronously (each thread copies 16B) ----
  const _Float16* Qb = qkv + (size_t)(b * T_ + q0) * D3_ + h * HD_;
  {
    uint32_t dst = lds_off(sQ) + (uint32_t)tid * 16u;              // byte offset in LDS
    uint64_t src = (uint64_t)(uintptr_t)(Qb + (size_t)(tid >> 3) * D3_ + ((tid & 7) << 3));
    asm volatile("global_load_async_to_lds_b128 %0, %1, off" :: "v"(dst), "v"(src) : "memory");
    asm volatile("s_wait_asynccnt 0" ::: "memory");
  }
  __syncthreads();

  // ---- Phase 1: S = (Q @ K^T) * scale, causal-masked, into LDS ----
  v16h qa0 = load_a_frag(sQ, HD_, 0);
  v16h qa1 = load_a_frag(sQ, HD_, 32);
  for (int jt = wave; jt < T_ / 16; jt += 4) {
    const _Float16* Kb = qkv + (size_t)(b * T_ + jt * 16) * D3_ + D_ + h * HD_;
    v16h kb0 = load_b_frag(Kb, D3_, 0);
    v16h kb1 = load_b_frag(Kb, D3_, 32);
    v8f c = {};
    c = wmma32(qa0, kb0, c);
    c = wmma32(qa1, kb1, c);
    int key = jt * 16 + (lane & 15);
#pragma unroll
    for (int r = 0; r < 8; ++r) {
      int m = ((lane >> 4) << 3) + r;
      float sv = c[r] * scale;
      if (key > q0 + m) sv = -3.0e38f;
      sS[m * T_ + key] = sv;
    }
  }
  __syncthreads();

  // ---- Phase 2: softmax (unnormalized exp in LDS; 1/sum kept per row) ----
  {
    int row = tid >> 3, sub = tid & 7;
    float* Sr = sS + row * T_ + sub * 128;
    float mx = -3.0e38f;
    for (int i = 0; i < 128; ++i) mx = fmaxf(mx, Sr[i]);
    sRed[row * 8 + sub] = mx;
    __syncthreads();
    if (sub == 0) {
      float m2 = sRed[row * 8];
      for (int i = 1; i < 8; ++i) m2 = fmaxf(m2, sRed[row * 8 + i]);
      sSt[row] = m2;
    }
    __syncthreads();
    float rm = sSt[row], sum = 0.f;
    for (int i = 0; i < 128; ++i) { float e = __expf(Sr[i] - rm); Sr[i] = e; sum += e; }
    sRed[row * 8 + sub] = sum;
    __syncthreads();
    if (sub == 0) {
      float tt = 0.f;
      for (int i = 0; i < 8; ++i) tt += sRed[row * 8 + i];
      sSt[16 + row] = 1.f / tt;
    }
    __syncthreads();
  }

  // ---- Phase 3: O = P @ V^T-layout (each wave covers 256 keys) ----
  const _Float16* Vt = vt + (size_t)(b * H_ + h) * HD_ * T_;   // row d, col t, ld = T
  v8f o0 = {}, o1 = {}, o2 = {}, o3 = {};
  for (int kk = wave * 256; kk < (wave + 1) * 256; kk += 32) {
    v16h pa = load_a_frag_lds_f32(sS, T_, kk);
    o0 = wmma32(pa, load_b_frag(Vt +  0 * T_, T_, kk), o0);
    o1 = wmma32(pa, load_b_frag(Vt + 16 * T_, T_, kk), o1);
    o2 = wmma32(pa, load_b_frag(Vt + 32 * T_, T_, kk), o2);
    o3 = wmma32(pa, load_b_frag(Vt + 48 * T_, T_, kk), o3);
  }
  v8f* os[4] = {&o0, &o1, &o2, &o3};
#pragma unroll
  for (int nt = 0; nt < 4; ++nt) {
    v8f ov = *os[nt];
#pragma unroll
    for (int r = 0; r < 8; ++r) {
      int m = ((lane >> 4) << 3) + r;
      sO[wave * (16 * 64) + m * 64 + nt * 16 + (lane & 15)] = ov[r];
    }
  }
  __syncthreads();
  for (int i = tid; i < 16 * 64; i += 128) {
    int m = i >> 6, col = i & 63;
    float v = sO[i] + sO[1024 + i] + sO[2048 + i] + sO[3072 + i];
    v *= sSt[16 + m];
    y16[(size_t)(b * T_ + q0 + m) * D_ + h * HD_ + col] = (_Float16)v;
  }
}

// ---------------------------------------------------------------------------
// LM head: one warp per (b, v); out[b,v] = dot(lnf_x16[b,:], wte[v,:])
// ---------------------------------------------------------------------------
__global__ __launch_bounds__(256) void lm_head_kernel(const _Float16* __restrict__ x16,
                                                      const float* __restrict__ wte,
                                                      float* __restrict__ out) {
  long gw = ((long)blockIdx.x * blockDim.x + threadIdx.x) >> 5;
  if (gw >= (long)B_ * V_) return;
  int lane = threadIdx.x & 31;
  int b = (int)(gw / V_), v = (int)(gw % V_);
  const _Float16* xr = x16 + (size_t)b * D_;
  const float* wr = wte + (size_t)v * D_;
  float s = 0.f;
  for (int k = lane; k < D_; k += 32) s += (float)xr[k] * wr[k];
#pragma unroll
  for (int off = 16; off > 0; off >>= 1) s += __shfl_down(s, off, 32);
  if (lane == 0) out[(size_t)b * V_ + v] = s;
}

// ---------------------------------------------------------------------------
// Host orchestration
// ---------------------------------------------------------------------------
extern "C" void kernel_launch(void* const* d_in, const int* in_sizes, int n_in,
                              void* d_out, int out_size, void* d_ws, size_t ws_size,
                              hipStream_t stream) {
  (void)in_sizes; (void)n_in; (void)out_size; (void)ws_size;
  const int*   idx    = (const int*)  d_in[0];
  const float* wte    = (const float*)d_in[1];
  const float* wpe    = (const float*)d_in[2];
  const float* ln1_w  = (const float*)d_in[3];
  const float* ln1_b  = (const float*)d_in[4];
  const float* attn_w = (const float*)d_in[5];
  const float* attn_b = (const float*)d_in[6];
  const float* proj_w = (const float*)d_in[7];
  const float* proj_b = (const float*)d_in[8];
  const float* ln2_w  = (const float*)d_in[9];
  const float* ln2_b  = (const float*)d_in[10];
  const float* fc_w   = (const float*)d_in[11];
  const float* fc_b   = (const float*)d_in[12];
  const float* fcp_w  = (const float*)d_in[13];
  const float* fcp_b  = (const float*)d_in[14];
  const float* lnf_w  = (const float*)d_in[15];
  const float* lnf_b  = (const float*)d_in[16];
  float* out = (float*)d_out;

  // Workspace carve-up (~104 MiB total)
  char* p = (char*)d_ws;
  float*    x     = (float*)p;     p += (size_t)NTOK * D_  * 4;  // residual stream f32
  _Float16* h16   = (_Float16*)p;  p += (size_t)NTOK * D_  * 2;  // LN output f16
  _Float16* qkv16 = (_Float16*)p;  p += (size_t)NTOK * D3_ * 2;  // qkv f16
  _Float16* y16   = (_Float16*)p;  p += (size_t)NTOK * D_  * 2;  // attn out f16
  _Float16* m16   = (_Float16*)p;  p += (size_t)NTOK * D4_ * 2;  // gelu(fc) f16
  _Float16* w16   = (_Float16*)p;  p += (size_t)D4_  * D_  * 2;  // per-GEMM weight f16
  _Float16* vt    = (_Float16*)p;  p += (size_t)B_ * H_ * HD_ * T_ * 2;  // V transposed

  auto gemm = [&](const _Float16* A, const _Float16* W, const float* bias,
                  float* outf, const float* resid, _Float16* outh,
                  int M, int N, int K, int gelu) {
    dim3 grid(N / 128, M / 128);
    wmma_gemm_kernel<<<grid, 256, 0, stream>>>(A, W, bias, outf, resid, outh, N, K, gelu);
  };
  auto cvt = [&](const float* src, _Float16* dst, long n) {
    cvt_f32_f16_kernel<<<1024, 256, 0, stream>>>(src, dst, n);
  };

  constexpr int ATT_SMEM = (16 * 1024 + 4 * 16 * 64 + 16 * 8 + 32) * (int)sizeof(float)
                         + 16 * 64 * 2;  // + f16 Q tile

  // x = wte[idx] + wpe
  embed_kernel<<<2048, 256, 0, stream>>>(idx, wte, wpe, x);

  for (int l = 0; l < L_; ++l) {
    // h = LN1(x) -> f16
    layernorm_f16_kernel<<<NTOK, 256, 0, stream>>>(x, ln1_w + (size_t)l * D_, ln1_b + (size_t)l * D_, h16, D_);
    // qkv16 = h @ attn_w^T + attn_b
    cvt(attn_w + (size_t)l * D3_ * D_, w16, (long)D3_ * D_);
    gemm(h16, w16, attn_b + (size_t)l * D3_, nullptr, nullptr, qkv16, NTOK, D3_, D_, 0);
    // vt = transpose(V)
    vtrans_kernel<<<2048, 256, 0, stream>>>(qkv16, vt);
    // y16 = causal softmax attention
    attn_kernel<<<dim3(T_ / 16, H_, B_), 128, ATT_SMEM, stream>>>(qkv16, vt, y16);
    // x += y @ proj_w^T + proj_b
    cvt(proj_w + (size_t)l * D_ * D_, w16, (long)D_ * D_);
    gemm(y16, w16, proj_b + (size_t)l * D_, x, x, nullptr, NTOK, D_, D_, 0);
    // h = LN2(x)
    layernorm_f16_kernel<<<NTOK, 256, 0, stream>>>(x, ln2_w + (size_t)l * D_, ln2_b + (size_t)l * D_, h16, D_);
    // m16 = gelu(h @ fc_w^T + fc_b)
    cvt(fc_w + (size_t)l * D4_ * D_, w16, (long)D4_ * D_);
    gemm(h16, w16, fc_b + (size_t)l * D4_, nullptr, nullptr, m16, NTOK, D4_, D_, 1);
    // x += m @ fcp_w^T + fcp_b
    cvt(fcp_w + (size_t)l * D_ * D4_, w16, (long)D_ * D4_);
    gemm(m16, w16, fcp_b + (size_t)l * D_, x, x, nullptr, NTOK, D_, D4_, 0);
  }

  // Final LN on the last position of each batch row only (4 rows)
  layernorm_f16_kernel<<<B_, 256, 0, stream>>>(x + (size_t)(T_ - 1) * D_, lnf_w, lnf_b, h16, (long)T_ * D_);

  // logits = lnf_x @ wte^T  (4 x 50257, warp per output)
  long warps = (long)B_ * V_;
  int lm_blocks = (int)((warps * 32 + 255) / 256);
  lm_head_kernel<<<lm_blocks, 256, 0, stream>>>(h16, wte, out);
}
